// EquivariantCrossAttention_33870112096298
// MI455X (gfx1250) — 16-bit and 8-bit matrix operand order repaired and checked by reading, compiled, not run on hardware
//
#include <hip/hip_runtime.h>
#include <math.h>


#define B_   2
#define C_   256
#define Z_   128
#define H_   64
#define NH_  8
#define HH_  512

typedef __attribute__((ext_vector_type(16))) __bf16 bfx16;
typedef __attribute__((ext_vector_type(8)))  __bf16 bfx8;
typedef __attribute__((ext_vector_type(8)))  float  fx8;
typedef __attribute__((ext_vector_type(16))) int    i32x16;

__device__ __forceinline__ unsigned short f2bf(float f) {
  unsigned u = __float_as_uint(f);
  u = u + 0x7FFFu + ((u >> 16) & 1u);
  return (unsigned short)(u >> 16);
}
__device__ __forceinline__ float bf2f(unsigned short h) {
  return __uint_as_float(((unsigned)h) << 16);
}

__device__ __forceinline__ unsigned char f2e4m3(float f) {
  unsigned u  = __float_as_uint(f);
  unsigned s  = (u >> 24) & 0x80u;
  unsigned au = u & 0x7FFFFFFFu;
  if (au >= 0x43E00000u) return (unsigned char)(s | 0x7Eu);
  if (au < 0x3C800000u) {
    int qi = (int)rintf(__uint_as_float(au) * 512.0f);
    if (qi > 8) qi = 8;
    return (unsigned char)(s | (unsigned)qi);
  }
  unsigned r = au + 0x7FFFFu + ((au >> 20) & 1u);
  if (r >= 0x43E00000u) return (unsigned char)(s | 0x7Eu);
  unsigned e4 = (r >> 23) - 120u;
  return (unsigned char)(s | (e4 << 3) | ((r >> 20) & 7u));
}
__device__ __forceinline__ float e4m3f(unsigned v) {
  unsigned e = (v >> 3) & 15u, m = v & 7u;
  float f = e ? __uint_as_float(((e + 120u) << 23) | (m << 20))
              : (float)m * 0.001953125f;
  return (v & 0x80u) ? -f : f;
}

__device__ __forceinline__ float gelu_tanh(float x) {
  const float k = 0.7978845608028654f;
  float t = tanhf(k * (x + 0.044715f * x * x * x));
  return 0.5f * x * (1.0f + t);
}

__device__ __forceinline__ bfx16 load_a_lds(const unsigned short* base, int stride,
                                            int k0, int lane) {
  int m = lane & 15, half = lane >> 4;
  const unsigned short* ptr = base + m * stride + k0 + half * 16;
  bfx8 lo = *(const bfx8*)(ptr);
  bfx8 hi = *(const bfx8*)(ptr + 8);
  return __builtin_shufflevector(lo, hi, 0,1,2,3,4,5,6,7,8,9,10,11,12,13,14,15);
}
__device__ __forceinline__ bfx16 load_b_glb(const unsigned short* Wt, int Kdim,
                                            int n0, int k0, int lane) {
  int n = n0 + (lane & 15), half = lane >> 4;
  return *(const bfx16*)(Wt + n * Kdim + k0 + half * 16);
}
__device__ __forceinline__ fx8 wmma_bf16(bfx16 a, bfx16 b, fx8 c) {
  return __builtin_amdgcn_wmma_f32_16x16x32_bf16(false, a, false, b, (short)0, c,
                                                 false, false);
}

__device__ __forceinline__ i32x16 load_a8_lds(const unsigned char* base, int stride,
                                              int k0, int lane) {
  int m = lane & 15, half = lane >> 4;
  const unsigned char* p = base + m * stride + k0 + half * 8;
  i32x16 a;
  #pragma unroll
  for (int c = 0; c < 8; ++c) {
    unsigned long long d = *(const unsigned long long*)(p + c * 16);
    a[2 * c]     = (int)(unsigned)(d & 0xFFFFFFFFull);
    a[2 * c + 1] = (int)(unsigned)(d >> 32);
  }
  return a;
}
__device__ __forceinline__ i32x16 load_b8_glb(const unsigned char* Wt, int Kdim,
                                              int n0, int k0, int lane) {
  int n = n0 + (lane & 15), half = lane >> 4;
  const unsigned char* p = Wt + n * Kdim + k0 + half * 8;
  i32x16 b;
  #pragma unroll
  for (int g = 0; g < 8; ++g) {
    unsigned long long d = *(const unsigned long long*)(p + g * 16);
    b[2 * g]     = (int)(unsigned)(d & 0xFFFFFFFFull);
    b[2 * g + 1] = (int)(unsigned)(d >> 32);
  }
  return b;
}
__device__ __forceinline__ fx8 wmma_fp8(i32x16 a, i32x16 b, fx8 c) {
  return __builtin_amdgcn_wmma_f32_16x16x128_fp8_fp8(a, b, (short)0, c, false, false);
}

__global__ void wcvt_kernel(const float* __restrict__ src,
                            unsigned short* __restrict__ dst, int K, int N) {
  int i = blockIdx.x * blockDim.x + threadIdx.x;
  if (i < K * N) { int k = i / N, n = i % N; dst[n * K + k] = f2bf(src[i]); }
}
__global__ void wcvt8_kernel(const float* __restrict__ src,
                             unsigned char* __restrict__ dst, int K, int N) {
  int i = blockIdx.x * blockDim.x + threadIdx.x;
  if (i < K * N) { int k = i / N, n = i % N; dst[n * K + k] = f2e4m3(src[i]); }
}
__global__ void kv_kernel(const float* __restrict__ a,
                          const float* __restrict__ Wk, const float* __restrict__ bk,
                          const float* __restrict__ Wv, const float* __restrict__ bv,
                          float* __restrict__ Kout, float* __restrict__ Vout) {
  int i = blockIdx.x * blockDim.x + threadIdx.x;
  int n  = i & (HH_ - 1);
  int zz = i >> 9;
  const float* arow = a + zz * H_;
  float sk = bk[n], sv = bv[n];
  #pragma unroll 4
  for (int k = 0; k < H_; ++k) {
    float av = arow[k];
    sk += av * Wk[k * HH_ + n];
    sv += av * Wv[k * HH_ + n];
  }
  Kout[i] = sk; Vout[i] = sv;
}

struct EqcaParams {
  const float *inputs, *p;
  const float *B_q, *bqe, *B_v, *bve, *bq;
  const float *vb1, *vg, *vbn, *vb2;
  const float *mb1, *mg, *mbn, *mb2;
  const float *Wo, *bo;
  const float *Kws, *V0ws;
  const unsigned short *Wqe_t, *Wve_t, *Wq_t, *vW1_t, *vW2_t;
  const unsigned char  *mW1_8, *mW2_8;
  float* out;
};

#define SF_S   72
#define VF_S   528
#define OFF_VF 0
#define OFF_HB 67584
#define ARENA  135168

__global__ __launch_bounds__(256) void eqca_main(EqcaParams P) {
  const int tid  = threadIdx.x;
  const int lane = tid & 31;
  const int w    = tid >> 5;
  const int half = lane >> 4;
  const int c    = blockIdx.x;
  const int b    = blockIdx.y;

  __shared__ alignas(16) unsigned char arena[ARENA];
  __shared__ float slog[Z_ * NH_];
  __shared__ float sy[HH_];
  __shared__ float srm[Z_], srr[Z_];
  __shared__ float sBm[96];
  __shared__ float sp[Z_ * 3];

  unsigned short* sF  = (unsigned short*)(arena + OFF_VF);
  unsigned short* sE  = (unsigned short*)(arena + OFF_HB);
  unsigned short* sH1 = (unsigned short*)(arena + OFF_HB + 18432);
  unsigned char*  sVf = arena + OFF_VF;
  unsigned char*  sHb = arena + OFF_HB;

  for (int i = tid; i < Z_ * NH_; i += 256) slog[i] = 0.f;
  for (int i = tid; i < HH_; i += 256) sy[i] = 0.f;
  for (int i = tid; i < Z_ * 3; i += 256) sp[i] = P.p[b * Z_ * 3 + i];
  if (tid < 96) sBm[tid] = P.B_q[tid];

  float xin0 = P.inputs[(b * C_ + c) * 3 + 0];
  float xin1 = P.inputs[(b * C_ + c) * 3 + 1];
  float xin2 = P.inputs[(b * C_ + c) * 3 + 2];
  const float TWO_PI = 6.283185307179586f;
  __syncthreads();

  for (int e = tid; e < Z_ * H_; e += 256) {
    int z = e >> 6, col = e & 63, j = col & 31;
    float i0 = xin0 - sp[z * 3 + 0];
    float i1 = xin1 - sp[z * 3 + 1];
    float i2 = xin2 - sp[z * 3 + 2];
    float pr = TWO_PI * (i0 * sBm[j] + i1 * sBm[32 + j] + i2 * sBm[64 + j]);
    sF[z * SF_S + col] = f2bf(col < 32 ? sinf(pr) : cosf(pr));
  }
  __syncthreads();
  #pragma unroll 1
  for (int i = 0; i < 4; ++i) {
    int t2 = w + 8 * i, mt = t2 >> 2, nt = t2 & 3;
    fx8 acc = {};
    #pragma unroll
    for (int ks = 0; ks < 2; ++ks) {
      bfx16 a  = load_a_lds(sF + mt * 16 * SF_S, SF_S, ks * 32, lane);
      bfx16 bb = load_b_glb(P.Wqe_t, 64, nt * 16, ks * 32, lane);
      acc = wmma_bf16(a, bb, acc);
    }
    int n = nt * 16 + (lane & 15);
    float bias = P.bqe[n];
    #pragma unroll
    for (int r = 0; r < 8; ++r)
      sE[(mt * 16 + r + 8 * half) * SF_S + n] = f2bf(acc[r] + bias);
  }
  __syncthreads();
  #pragma unroll 1
  for (int mt = 0; mt < 8; ++mt) {
    float dotv[8];
    #pragma unroll
    for (int r = 0; r < 8; ++r) dotv[r] = 0.f;
    #pragma unroll 1
    for (int t = 0; t < 4; ++t) {
      int nt = 4 * w + t;
      __builtin_prefetch(P.Wq_t + (nt * 16) * 64, 0, 1);
      fx8 acc = {};
      #pragma unroll
      for (int ks = 0; ks < 2; ++ks) {
        bfx16 a  = load_a_lds(sE + mt * 16 * SF_S, SF_S, ks * 32, lane);
        bfx16 bb = load_b_glb(P.Wq_t, 64, nt * 16, ks * 32, lane);
        acc = wmma_bf16(a, bb, acc);
      }
      int n = nt * 16 + (lane & 15);
      float bias = P.bq[n];
      #pragma unroll
      for (int r = 0; r < 8; ++r) {
        int z = mt * 16 + r + 8 * half;
        dotv[r] += (acc[r] + bias) * P.Kws[(b * Z_ + z) * HH_ + n];
      }
    }
    #pragma unroll
    for (int r = 0; r < 8; ++r)
      atomicAdd(&slog[(mt * 16 + r + 8 * half) * NH_ + w], dotv[r]);
  }
  __syncthreads();
  if (tid < NH_) {
    const float scale = 0.125f;
    float mx = -3.4e38f;
    for (int z = 0; z < Z_; ++z) mx = fmaxf(mx, slog[z * NH_ + tid] * scale);
    float s = 0.f;
    for (int z = 0; z < Z_; ++z) {
      float e = expf(slog[z * NH_ + tid] * scale - mx);
      slog[z * NH_ + tid] = e; s += e;
    }
    float inv = 1.f / s;
    for (int z = 0; z < Z_; ++z) slog[z * NH_ + tid] *= inv;
  }
  if (tid < 96) sBm[tid] = P.B_v[tid];
  __syncthreads();

  for (int e = tid; e < Z_ * H_; e += 256) {
    int z = e >> 6, col = e & 63, j = col & 31;
    float i0 = xin0 - sp[z * 3 + 0];
    float i1 = xin1 - sp[z * 3 + 1];
    float i2 = xin2 - sp[z * 3 + 2];
    float pr = TWO_PI * (i0 * sBm[j] + i1 * sBm[32 + j] + i2 * sBm[64 + j]);
    sF[z * SF_S + col] = f2bf(col < 32 ? sinf(pr) : cosf(pr));
  }
  __syncthreads();
  #pragma unroll 1
  for (int i = 0; i < 4; ++i) {
    int t2 = w + 8 * i, mt = t2 >> 2, nt = t2 & 3;
    fx8 acc = {};
    #pragma unroll
    for (int ks = 0; ks < 2; ++ks) {
      bfx16 a  = load_a_lds(sF + mt * 16 * SF_S, SF_S, ks * 32, lane);
      bfx16 bb = load_b_glb(P.Wve_t, 64, nt * 16, ks * 32, lane);
      acc = wmma_bf16(a, bb, acc);
    }
    int n = nt * 16 + (lane & 15);
    float bias = P.bve[n];
    #pragma unroll
    for (int r = 0; r < 8; ++r)
      sE[(mt * 16 + r + 8 * half) * SF_S + n] = f2bf(acc[r] + bias);
  }
  __syncthreads();
  #pragma unroll 1
  for (int i = 0; i < 4; ++i) {
    int t2 = w + 8 * i, mt = t2 >> 2, nt = t2 & 3;
    fx8 acc = {};
    #pragma unroll
    for (int ks = 0; ks < 2; ++ks) {
      bfx16 a  = load_a_lds(sE + mt * 16 * SF_S, SF_S, ks * 32, lane);
      bfx16 bb = load_b_glb(P.vW1_t, 64, nt * 16, ks * 32, lane);
      acc = wmma_bf16(a, bb, acc);
    }
    int n = nt * 16 + (lane & 15);
    float bias = P.vb1[n];
    #pragma unroll
    for (int r = 0; r < 8; ++r)
      sH1[(mt * 16 + r + 8 * half) * SF_S + n] = f2bf(gelu_tanh(acc[r] + bias));
  }
  __syncthreads();
  if (tid < Z_) {
    float s = 0.f, s2 = 0.f;
    for (int k = 0; k < H_; ++k) { float x = bf2f(sH1[tid * SF_S + k]); s += x; s2 += x * x; }
    float m = s * (1.f / H_), v = s2 * (1.f / H_) - m * m;
    srm[tid] = m; srr[tid] = rsqrtf(v + 1e-5f);
  }
  __syncthreads();
  for (int e = tid; e < Z_ * H_; e += 256) {
    int z = e >> 6, col = e & 63;
    float x = bf2f(sH1[z * SF_S + col]);
    sH1[z * SF_S + col] = f2bf((x - srm[z]) * srr[z] * P.vg[col] + P.vbn[col]);
  }
  __syncthreads();
  #pragma unroll 1
  for (int mt = 0; mt < 8; ++mt) {
    #pragma unroll 1
    for (int i = 0; i < 4; ++i) {
      int nt = w + 8 * i;
      __builtin_prefetch(P.vW2_t + (nt * 16) * 64, 0, 1);
      fx8 ag = {}, ab = {};
      #pragma unroll
      for (int ks = 0; ks < 2; ++ks) {
        bfx16 a  = load_a_lds(sH1 + mt * 16 * SF_S, SF_S, ks * 32, lane);
        bfx16 bg = load_b_glb(P.vW2_t, 64, nt * 16, ks * 32, lane);
        bfx16 bb = load_b_glb(P.vW2_t, 64, 512 + nt * 16, ks * 32, lane);
        ag = wmma_bf16(a, bg, ag);
        ab = wmma_bf16(a, bb, ab);
      }
      int n = nt * 16 + (lane & 15);
      float biasg = P.vb2[n], biasb = P.vb2[512 + n];
      #pragma unroll
      for (int r = 0; r < 8; ++r) {
        int z = mt * 16 + r + 8 * half;
        float v0 = P.V0ws[(b * Z_ + z) * HH_ + n];
        sVf[z * VF_S + n] = f2e4m3(v0 * (1.f + ag[r] + biasg) + (ab[r] + biasb));
      }
    }
  }
  __syncthreads();
  #pragma unroll 1
  for (int mt = 0; mt < 8; ++mt) {
    #pragma unroll 1
    for (int i = 0; i < 4; ++i) {
      int nt = w + 8 * i;
      __builtin_prefetch(P.mW1_8 + (nt * 16) * 512, 0, 1);
      fx8 acc = {};
      #pragma unroll
      for (int ks = 0; ks < 4; ++ks) {
        i32x16 a8 = load_a8_lds(sVf + mt * 16 * VF_S, VF_S, ks * 128, lane);
        i32x16 b8 = load_b8_glb(P.mW1_8, 512, nt * 16, ks * 128, lane);
        acc = wmma_fp8(a8, b8, acc);
      }
      int n = nt * 16 + (lane & 15);
      float bias = P.mb1[n];
      #pragma unroll
      for (int r = 0; r < 8; ++r)
        sHb[(mt * 16 + r + 8 * half) * VF_S + n] = f2e4m3(gelu_tanh(acc[r] + bias));
    }
  }
  __syncthreads();
  if (tid < Z_) {
    float s = 0.f, s2 = 0.f;
    for (int k = 0; k < HH_; ++k) { float x = e4m3f(sHb[tid * VF_S + k]); s += x; s2 += x * x; }
    float m = s * (1.f / HH_), v = s2 * (1.f / HH_) - m * m;
    srm[tid] = m; srr[tid] = rsqrtf(v + 1e-5f);
  }
  __syncthreads();
  for (int e = tid; e < Z_ * HH_; e += 256) {
    int z = e >> 9, col = e & 511;
    float x = e4m3f(sHb[z * VF_S + col]);
    sHb[z * VF_S + col] = f2e4m3((x - srm[z]) * srr[z] * P.mg[col] + P.mbn[col]);
  }
  __syncthreads();
  #pragma unroll 1
  for (int mt = 0; mt < 8; ++mt) {
    #pragma unroll 1
    for (int i = 0; i < 4; ++i) {
      int nt = w + 8 * i;
      __builtin_prefetch(P.mW2_8 + (nt * 16) * 512, 0, 1);
      fx8 acc = {};
      #pragma unroll
      for (int ks = 0; ks < 4; ++ks) {
        i32x16 a8 = load_a8_lds(sHb + mt * 16 * VF_S, VF_S, ks * 128, lane);
        i32x16 b8 = load_b8_glb(P.mW2_8, 512, nt * 16, ks * 128, lane);
        acc = wmma_fp8(a8, b8, acc);
      }
      int n = nt * 16 + (lane & 15);
      int h = n >> 6;
      float bias = P.mb2[n];
      float s = 0.f;
      #pragma unroll
      for (int r = 0; r < 8; ++r) {
        int z = mt * 16 + r + 8 * half;
        s += slog[z * NH_ + h] * (acc[r] + bias);
      }
      atomicAdd(&sy[n], s);
    }
  }
  __syncthreads();
  for (int n = tid; n < HH_; n += 256) {
    float acc = P.bo[n];
    #pragma unroll 4
    for (int k = 0; k < HH_; ++k) acc += sy[k] * P.Wo[k * HH_ + n];
    P.out[(b * C_ + c) * HH_ + n] = acc;
  }
}

extern "C" void kernel_launch(void* const* d_in, const int* in_sizes, int n_in,
                              void* d_out, int out_size, void* d_ws, size_t ws_size,
                              hipStream_t stream) {
  (void)in_sizes; (void)n_in; (void)out_size; (void)ws_size;
  const float* a   = (const float*)d_in[2];
  const float* Wqe = (const float*)d_in[4];
  const float* Wve = (const float*)d_in[7];
  const float* Wq  = (const float*)d_in[9];
  const float* Wk  = (const float*)d_in[11];
  const float* bk  = (const float*)d_in[12];
  const float* Wv  = (const float*)d_in[13];
  const float* bv  = (const float*)d_in[14];
  const float* vW1 = (const float*)d_in[15];
  const float* vW2 = (const float*)d_in[19];
  const float* mW1 = (const float*)d_in[21];
  const float* mW2 = (const float*)d_in[25];

  char* ws = (char*)d_ws;
  float*          Kws   = (float*)(ws + 0);
  float*          V0ws  = (float*)(ws + 524288);
  unsigned short* Wqe_t = (unsigned short*)(ws + 1048576);
  unsigned short* Wve_t = (unsigned short*)(ws + 1056768);
  unsigned short* Wq_t  = (unsigned short*)(ws + 1064960);
  unsigned short* vW1_t = (unsigned short*)(ws + 1130496);
  unsigned short* vW2_t = (unsigned short*)(ws + 1138688);
  unsigned char*  mW1_8 = (unsigned char*)(ws + 1269760);
  unsigned char*  mW2_8 = (unsigned char*)(ws + 1531904);

  auto cvt = [&](const float* src, unsigned short* dst, int K, int N) {
    wcvt_kernel<<<(K * N + 255) / 256, 256, 0, stream>>>(src, dst, K, N);
  };
  cvt(Wqe, Wqe_t, 64, 64);
  cvt(Wve, Wve_t, 64, 64);
  cvt(Wq,  Wq_t,  64, 512);
  cvt(vW1, vW1_t, 64, 64);
  cvt(vW2, vW2_t, 64, 1024);
  wcvt8_kernel<<<(512 * 512 + 255) / 256, 256, 0, stream>>>(mW1, mW1_8, 512, 512);
  wcvt8_kernel<<<(512 * 512 + 255) / 256, 256, 0, stream>>>(mW2, mW2_8, 512, 512);

  kv_kernel<<<(B_ * Z_ * HH_) / 256, 256, 0, stream>>>(a, Wk, bk, Wv, bv, Kws, V0ws);

  EqcaParams P;
  P.inputs = (const float*)d_in[0];
  P.p      = (const float*)d_in[1];
  P.B_q    = (const float*)d_in[3];
  P.bqe    = (const float*)d_in[5];
  P.B_v    = (const float*)d_in[6];
  P.bve    = (const float*)d_in[8];
  P.bq     = (const float*)d_in[10];
  P.vb1    = (const float*)d_in[16];
  P.vg     = (const float*)d_in[17];
  P.vbn    = (const float*)d_in[18];
  P.vb2    = (const float*)d_in[20];
  P.mb1    = (const float*)d_in[22];
  P.mg     = (const float*)d_in[23];
  P.mbn    = (const float*)d_in[24];
  P.mb2    = (const float*)d_in[26];
  P.Wo     = (const float*)d_in[27];
  P.bo     = (const float*)d_in[28];
  P.Kws = Kws; P.V0ws = V0ws;
  P.Wqe_t = Wqe_t; P.Wve_t = Wve_t; P.Wq_t = Wq_t;
  P.vW1_t = vW1_t; P.vW2_t = vW2_t;
  P.mW1_8 = mW1_8; P.mW2_8 = mW2_8;
  P.out = (float*)d_out;

  dim3 grid(C_, B_);
  eqca_main<<<grid, 256, 0, stream>>>(P);
}
